// Self_Attentnion_v3_1159641170680
// MI455X (gfx1250) — matrix operand order repaired and checked by reading, not compiled, not run on hardware
//
#include <hip/hip_runtime.h>
#include <hip/hip_bf16.h>

// ---------------------------------------------------------------------------
// Self-attention (single head, d=1024) for MI455X / gfx1250.
// All GEMMs run on v_wmma_f32_16x16x32_bf16 (bf16 inputs, fp32 accumulate).
// Mainloop uses GLOBAL_LOAD_ASYNC_TO_LDS_B128 + double-buffered LDS when the
// toolchain exposes the async builtins; otherwise falls back to the manual
// global->VGPR->LDS staging path.
// ---------------------------------------------------------------------------

typedef __bf16 bf16;
typedef __attribute__((ext_vector_type(16))) __bf16 v16bf;
typedef __attribute__((ext_vector_type(8)))  __bf16 v8bf;
typedef __attribute__((ext_vector_type(4)))  __bf16 v4bf;
typedef __attribute__((ext_vector_type(8)))  float  v8f;
typedef __attribute__((ext_vector_type(4)))  float  v4f;
typedef __attribute__((ext_vector_type(4)))  int    v4i;

#define HID   1024
#define SEQ   2048
#define BATCH 4
#define ROWS  (BATCH * SEQ)          // 8192
#define LDS_STRIDE 40                // bf16 elems per LDS row (32 + pad, 80B)

#if defined(__has_builtin)
#  if __has_builtin(__builtin_amdgcn_global_load_async_to_lds_b128)
#    define HAVE_ASYNC_LDS 1
#  endif
#endif
#ifndef HAVE_ASYNC_LDS
#  define HAVE_ASYNC_LDS 0
#endif

#if defined(__has_builtin) && __has_builtin(__builtin_amdgcn_s_wait_asynccnt)
#  define WAIT_ASYNC(N) __builtin_amdgcn_s_wait_asynccnt(N)
#else
#  define WAIT_ASYNC(N) asm volatile("s_wait_asynccnt " #N ::: "memory")
#endif

#define AS_GLOBAL __attribute__((address_space(1)))
#define AS_LOCAL  __attribute__((address_space(3)))

__device__ __forceinline__ bf16 f2bf(float f) {
  union { float f; unsigned u; } x; x.f = f;
  unsigned r = (x.u + 0x7FFFu + ((x.u >> 16) & 1u)) >> 16;
  unsigned short s = (unsigned short)r;
  bf16 b; __builtin_memcpy(&b, &s, sizeof(s));
  return b;
}

struct GemmSmem {
  __align__(16) bf16 a[128 * LDS_STRIDE];
  __align__(16) bf16 b[128 * LDS_STRIDE];
};

// ---- one K-step of WMMA compute from staged LDS panels ----
// A fragment: lane holds row; elems 0..7 -> K=8h..8h+7, elems 8..15 ->
// K=8h+16..8h+23 (ISA 16-bit A 16x32 layout).
// B fragment: lane holds column; same K layout as A.
__device__ __forceinline__ void compute_step(
    const bf16* Ash, const bf16* Bsh,
    int wm, int wn, int r, int half, v8f acc[4][2])
{
  v16bf af[4], bfg[2];
#pragma unroll
  for (int mi = 0; mi < 4; ++mi) {
    const bf16* base = Ash + (wm * 64 + mi * 16 + r) * LDS_STRIDE;
    v8bf lo = *(const v8bf*)(base + 8 * half);
    v8bf hi = *(const v8bf*)(base + 8 * half + 16);
#pragma unroll
    for (int e = 0; e < 8; ++e) { af[mi][e] = lo[e]; af[mi][e + 8] = hi[e]; }
  }
#pragma unroll
  for (int ni = 0; ni < 2; ++ni) {
    const bf16* base = Bsh + (wn * 32 + ni * 16 + r) * LDS_STRIDE;
    v8bf lo = *(const v8bf*)(base + 8 * half);
    v8bf hi = *(const v8bf*)(base + 8 * half + 16);
#pragma unroll
    for (int e = 0; e < 8; ++e) { bfg[ni][e] = lo[e]; bfg[ni][e + 8] = hi[e]; }
  }
#pragma unroll
  for (int mi = 0; mi < 4; ++mi)
#pragma unroll
    for (int ni = 0; ni < 2; ++ni)
      acc[mi][ni] = __builtin_amdgcn_wmma_f32_16x16x32_bf16(
          false, af[mi], false, bfg[ni], (short)0, acc[mi][ni], false, false);
}

// Stage 128x32 bf16 panels of A and B into LDS.
// Async path: 4 GLOBAL_LOAD_ASYNC_TO_LDS_B128 per wave per stage.
__device__ __forceinline__ void stage_panels(
    const bf16* __restrict__ Ablk, int lda,
    const bf16* __restrict__ Bblk, int ldb, int k0,
    bf16* Ash, bf16* Bsh, int tid)
{
#pragma unroll
  for (int i = 0; i < 2; ++i) {
    int c    = tid + i * 256;
    int row  = c >> 2;          // 4 chunks of 8 bf16 per 32-wide row
    int coff = (c & 3) << 3;
    const bf16* ga = Ablk + (size_t)row * lda + k0 + coff;
    const bf16* gb = Bblk + (size_t)row * ldb + k0 + coff;
    bf16* la = Ash + row * LDS_STRIDE + coff;
    bf16* lb = Bsh + row * LDS_STRIDE + coff;
#if HAVE_ASYNC_LDS
    __builtin_amdgcn_global_load_async_to_lds_b128(
        (AS_GLOBAL v4i*)ga, (AS_LOCAL v4i*)la, 0, 0);
    __builtin_amdgcn_global_load_async_to_lds_b128(
        (AS_GLOBAL v4i*)gb, (AS_LOCAL v4i*)lb, 0, 0);
#else
    *(v8bf*)la = *(const v8bf*)ga;
    *(v8bf*)lb = *(const v8bf*)gb;
#endif
  }
}

// 128x128 block tile, 256 threads = 8 waves in a 2(m) x 4(n) grid.
// Each wave: 64x32 tile = 4x2 fragments of 16x16, K stepped by 32.
// A: [M][K] bf16 row-major (K contiguous). B: [N][K] bf16 row-major.
__device__ __forceinline__ void gemm128(
    const bf16* __restrict__ Ablk, int lda,
    const bf16* __restrict__ Bblk, int ldb,
    int K, GemmSmem* sm, v8f acc[4][2])
{
  const int tid  = threadIdx.x;
  const int lane = tid & 31;
  const int wid  = tid >> 5;
  const int wm   = wid >> 2;   // 0..1
  const int wn   = wid & 3;    // 0..3
  const int r    = lane & 15;
  const int half = lane >> 4;
  const int nk   = K >> 5;

#if HAVE_ASYNC_LDS
  // Double-buffered pipeline: prefetch k+1 while computing k.
  stage_panels(Ablk, lda, Bblk, ldb, 0, sm[0].a, sm[0].b, tid);
  for (int kt = 0; kt < nk; ++kt) {
    if (kt + 1 < nk) {
      // Safe to overwrite buf[(kt+1)&1]: its last readers finished at the
      // closing barrier of iteration kt-1.
      stage_panels(Ablk, lda, Bblk, ldb, (kt + 1) << 5,
                   sm[(kt + 1) & 1].a, sm[(kt + 1) & 1].b, tid);
      WAIT_ASYNC(4);   // async loads retire in order: oldest 4 = stage kt
    } else {
      WAIT_ASYNC(0);
    }
    __syncthreads();
    compute_step(sm[kt & 1].a, sm[kt & 1].b, wm, wn, r, half, acc);
    __syncthreads();
  }
#else
  for (int kt = 0; kt < nk; ++kt) {
    stage_panels(Ablk, lda, Bblk, ldb, kt << 5, sm[0].a, sm[0].b, tid);
    __syncthreads();
    compute_step(sm[0].a, sm[0].b, wm, wn, r, half, acc);
    __syncthreads();
  }
#endif
}

#define GEMM_EPILOGUE_IDX                                     \
  const int lane = threadIdx.x & 31;                          \
  const int wid  = threadIdx.x >> 5;                          \
  const int wm   = wid >> 2;                                  \
  const int wn   = wid & 3;                                   \
  const int r    = lane & 15;                                 \
  const int half = lane >> 4;

#define ACC_INIT                                              \
  v8f acc[4][2];                                              \
  {                                                           \
    v8f z = {0.f,0.f,0.f,0.f,0.f,0.f,0.f,0.f};                \
    _Pragma("unroll")                                         \
    for (int i = 0; i < 4; ++i) { acc[i][0] = z; acc[i][1] = z; } \
  }

// ------------------------- elementwise / transpose -------------------------

__global__ void __launch_bounds__(256) k_convert_bf16(
    const float* __restrict__ x, bf16* __restrict__ y)
{
  size_t i = (size_t)blockIdx.x * blockDim.x + threadIdx.x;   // one float4 each
  v4f v = ((const v4f*)x)[i];
  v4bf o;
#pragma unroll
  for (int e = 0; e < 4; ++e) o[e] = f2bf(v[e]);
  *(v4bf*)(y + i * 4) = o;
}

// W[K][N] f32 -> WT[N][K] bf16 (32x32 tiles via LDS)
__global__ void __launch_bounds__(256) k_transpose_bf16(
    const float* __restrict__ W, bf16* __restrict__ WT, int K, int N)
{
  __shared__ float tile[32][33];
  int bx = blockIdx.x * 32;           // N
  int by = blockIdx.y * 32;           // K
  int tx = threadIdx.x & 31, ty = threadIdx.x >> 5;
#pragma unroll
  for (int i = 0; i < 32; i += 8)
    tile[ty + i][tx] = W[(size_t)(by + ty + i) * N + bx + tx];
  __syncthreads();
#pragma unroll
  for (int i = 0; i < 32; i += 8)
    WT[(size_t)(bx + ty + i) * K + by + tx] = f2bf(tile[tx][ty + i]);
}

// ------------------------------ GEMM kernels -------------------------------

// qkv = X @ Wqkv + bqkv ; scatter to Q (pre-scaled by 1/sqrt(d)), K, V^T
__global__ void __launch_bounds__(256) k_gemm_qkv(
    const bf16* __restrict__ Xbf, const bf16* __restrict__ WTqkv,
    const float* __restrict__ bqkv,
    bf16* __restrict__ Qbf, bf16* __restrict__ Kbf, bf16* __restrict__ VT)
{
  __shared__ GemmSmem sm[2];
  ACC_INIT
  const int row0 = blockIdx.y * 128;
  const int col0 = blockIdx.x * 128;
  gemm128(Xbf + (size_t)row0 * HID, HID,
          WTqkv + (size_t)col0 * HID, HID, HID, sm, acc);
  GEMM_EPILOGUE_IDX
#pragma unroll
  for (int mi = 0; mi < 4; ++mi)
#pragma unroll
    for (int ni = 0; ni < 2; ++ni)
#pragma unroll
      for (int e = 0; e < 8; ++e) {
        int row = row0 + wm * 64 + mi * 16 + e + half * 8;
        int col = col0 + wn * 32 + ni * 16 + r;
        float v = acc[mi][ni][e] + bqkv[col];
        int b = row >> 11, n = row & (SEQ - 1);
        if (col < HID)
          Qbf[(size_t)row * HID + col] = f2bf(v * 0.03125f);     // 1/sqrt(1024)
        else if (col < 2 * HID)
          Kbf[(size_t)row * HID + (col - HID)] = f2bf(v);
        else
          VT[((size_t)b * HID + (col - 2 * HID)) * SEQ + n] = f2bf(v);
      }
}

// S[b] = Q[b] @ K[b]^T  (scale already folded into Q)
__global__ void __launch_bounds__(256) k_gemm_scores(
    const bf16* __restrict__ Qbf, const bf16* __restrict__ Kbf,
    float* __restrict__ S)
{
  __shared__ GemmSmem sm[2];
  ACC_INIT
  const int b = blockIdx.z;
  const int row0 = blockIdx.y * 128;
  const int col0 = blockIdx.x * 128;
  gemm128(Qbf + ((size_t)b * SEQ + row0) * HID, HID,
          Kbf + ((size_t)b * SEQ + col0) * HID, HID, HID, sm, acc);
  GEMM_EPILOGUE_IDX
#pragma unroll
  for (int mi = 0; mi < 4; ++mi)
#pragma unroll
    for (int ni = 0; ni < 2; ++ni)
#pragma unroll
      for (int e = 0; e < 8; ++e) {
        int row = row0 + wm * 64 + mi * 16 + e + half * 8;
        int col = col0 + wn * 32 + ni * 16 + r;
        S[((size_t)b * SEQ + row) * SEQ + col] = acc[mi][ni][e];
      }
}

// Row softmax over 2048 fp32 scores; writes bf16 probs in place over the row
// (bf16 row stride 4096 elems == fp32 row stride 2048 elems == 8KB).
__global__ void __launch_bounds__(256) k_softmax(
    const float* __restrict__ S, bf16* __restrict__ P)
{
  __shared__ float red[256];
  const int row = blockIdx.x;
  const int tid = threadIdx.x;
  const float* s = S + (size_t)row * SEQ;
  bf16* p = P + (size_t)row * (2 * SEQ);
  float xs[8];
#pragma unroll
  for (int i = 0; i < 8; ++i) xs[i] = s[tid + i * 256];
  float m = -1e30f;
#pragma unroll
  for (int i = 0; i < 8; ++i) m = fmaxf(m, xs[i]);
  red[tid] = m; __syncthreads();
  for (int off = 128; off > 0; off >>= 1) {
    if (tid < off) red[tid] = fmaxf(red[tid], red[tid + off]);
    __syncthreads();
  }
  m = red[0]; __syncthreads();
  float sum = 0.f;
#pragma unroll
  for (int i = 0; i < 8; ++i) { xs[i] = __expf(xs[i] - m); sum += xs[i]; }
  red[tid] = sum; __syncthreads();
  for (int off = 128; off > 0; off >>= 1) {
    if (tid < off) red[tid] += red[tid + off];
    __syncthreads();
  }
  float inv = 1.0f / red[0];
#pragma unroll
  for (int i = 0; i < 8; ++i) p[tid + i * 256] = f2bf(xs[i] * inv);
}

// O[b] = P[b] @ V[b]   (V stored transposed: VT[b][d][n], n contiguous)
__global__ void __launch_bounds__(256) k_gemm_av(
    const bf16* __restrict__ P, const bf16* __restrict__ VT,
    bf16* __restrict__ Obf)
{
  __shared__ GemmSmem sm[2];
  ACC_INIT
  const int b = blockIdx.z;
  const int row0 = blockIdx.y * 128;
  const int col0 = blockIdx.x * 128;
  gemm128(P + ((size_t)b * SEQ + row0) * (2 * SEQ), 2 * SEQ,
          VT + ((size_t)b * HID + col0) * SEQ, SEQ, SEQ, sm, acc);
  GEMM_EPILOGUE_IDX
#pragma unroll
  for (int mi = 0; mi < 4; ++mi)
#pragma unroll
    for (int ni = 0; ni < 2; ++ni)
#pragma unroll
      for (int e = 0; e < 8; ++e) {
        int row = row0 + wm * 64 + mi * 16 + e + half * 8;
        int col = col0 + wn * 32 + ni * 16 + r;
        Obf[((size_t)b * SEQ + row) * HID + col] = f2bf(acc[mi][ni][e]);
      }
}

// out = O @ Wo + bo  (fp32 output)
__global__ void __launch_bounds__(256) k_gemm_out(
    const bf16* __restrict__ Obf, const bf16* __restrict__ WTo,
    const float* __restrict__ bo, float* __restrict__ out)
{
  __shared__ GemmSmem sm[2];
  ACC_INIT
  const int row0 = blockIdx.y * 128;
  const int col0 = blockIdx.x * 128;
  gemm128(Obf + (size_t)row0 * HID, HID,
          WTo + (size_t)col0 * HID, HID, HID, sm, acc);
  GEMM_EPILOGUE_IDX
#pragma unroll
  for (int mi = 0; mi < 4; ++mi)
#pragma unroll
    for (int ni = 0; ni < 2; ++ni)
#pragma unroll
      for (int e = 0; e < 8; ++e) {
        int row = row0 + wm * 64 + mi * 16 + e + half * 8;
        int col = col0 + wn * 32 + ni * 16 + r;
        out[(size_t)row * HID + col] = acc[mi][ni][e] + bo[col];
      }
}

// --------------------------------- launch ----------------------------------

extern "C" void kernel_launch(void* const* d_in, const int* in_sizes, int n_in,
                              void* d_out, int out_size, void* d_ws, size_t ws_size,
                              hipStream_t stream) {
  (void)in_sizes; (void)n_in; (void)out_size; (void)ws_size;
  const float* X    = (const float*)d_in[0];
  const float* Wqkv = (const float*)d_in[1];
  const float* bqkv = (const float*)d_in[2];
  const float* Wo   = (const float*)d_in[3];
  const float* bo   = (const float*)d_in[4];
  float* out = (float*)d_out;

  // Workspace carve-out (256B aligned), ~136 MB total.
  char* ws = (char*)d_ws;
  size_t off = 0;
  auto take = [&](size_t bytes) { char* p = ws + off; off += (bytes + 255) & ~(size_t)255; return p; };
  bf16* Xbf   = (bf16*)take((size_t)ROWS * HID * 2);        // 16 MB (reused as Obf)
  bf16* WTqkv = (bf16*)take((size_t)3 * HID * HID * 2);     //  6 MB
  bf16* WTo   = (bf16*)take((size_t)HID * HID * 2);         //  2 MB
  bf16* Qbf   = (bf16*)take((size_t)ROWS * HID * 2);        // 16 MB
  bf16* Kbf   = (bf16*)take((size_t)ROWS * HID * 2);        // 16 MB
  bf16* VT    = (bf16*)take((size_t)BATCH * HID * SEQ * 2); // 16 MB
  float* S    = (float*)take((size_t)BATCH * SEQ * SEQ * 4);// 64 MB
  bf16* P     = (bf16*)S;                                   // in-place overlay
  bf16* Obf   = Xbf;                                        // X no longer needed

  // 1) fp32 -> bf16 conversions / weight transposes
  k_convert_bf16<<<dim3((ROWS * HID) / (256 * 4)), 256, 0, stream>>>(X, Xbf);
  k_transpose_bf16<<<dim3(3 * HID / 32, HID / 32), 256, 0, stream>>>(Wqkv, WTqkv, HID, 3 * HID);
  k_transpose_bf16<<<dim3(HID / 32, HID / 32), 256, 0, stream>>>(Wo, WTo, HID, HID);
  // 2) QKV projection (+bias, Q pre-scaled, V transposed)
  k_gemm_qkv<<<dim3(3 * HID / 128, ROWS / 128), 256, 0, stream>>>(Xbf, WTqkv, bqkv, Qbf, Kbf, VT);
  // 3) scores = Q K^T
  k_gemm_scores<<<dim3(SEQ / 128, SEQ / 128, BATCH), 256, 0, stream>>>(Qbf, Kbf, S);
  // 4) softmax rows -> bf16 P
  k_softmax<<<dim3(BATCH * SEQ), 256, 0, stream>>>(S, P);
  // 5) O = P V
  k_gemm_av<<<dim3(HID / 128, SEQ / 128, BATCH), 256, 0, stream>>>(P, VT, Obf);
  // 6) out = O Wo + bo
  k_gemm_out<<<dim3(HID / 128, ROWS / 128), 256, 0, stream>>>(Obf, WTo, bo, out);
}
